// DualStreamAttention_63024350101861
// MI455X (gfx1250) — matrix operand order repaired and checked by reading, not compiled, not run on hardware
//
#include <hip/hip_runtime.h>
#include <hip/hip_bf16.h>

// ---------------------------------------------------------------------------
// DualStreamAttention for MI455X (gfx1250, wave32, WMMA).
// bf16 WMMA (f32 accum) for all matmuls; fused flash-style dual-softmax
// attention (QK^T computed once, never spilled to HBM); async-to-LDS loads
// (ASYNCcnt) for K/A tiles; global_prefetch for the log_mask HBM stream.
// ---------------------------------------------------------------------------

#define B_  2
#define T_  2048
#define D_  1024
#define H_  16
#define DH_ 64

typedef __bf16 bf16_t;
typedef __attribute__((ext_vector_type(16))) __bf16 v16bf;
typedef __attribute__((ext_vector_type(8)))  float  v8f;

static __device__ inline v8f zero_v8f() {
    v8f z;
#pragma unroll
    for (int i = 0; i < 8; ++i) z[i] = 0.0f;
    return z;
}

// Async DMA: copy 16 contiguous global bytes to a per-lane LDS address.
// Tracked by ASYNCcnt; drain with s_wait_asynccnt before consuming.
static __device__ inline void async_load_b128(void* lds, const void* gaddr) {
    unsigned l = (unsigned)(uintptr_t)lds;   // low 32 bits of generic = LDS addr
    asm volatile("global_load_async_to_lds_b128 %0, %1, off"
                 :: "v"(l), "v"(gaddr) : "memory");
}
static __device__ inline void wait_async0() {
    asm volatile("s_wait_asynccnt 0" ::: "memory");
}

// A-matrix fragment (16x32 bf16, MxK). Per CDNA5 ISA:
// lane L (L<16): row M=L, K pairs {0,1},{2,3},{4,5},{6,7},{16,17},...,{22,23}
// lane L (L>=16): row M=L-16, same pattern +8.
static __device__ inline v16bf load_a_frag(const bf16_t* __restrict__ base, int ldr) {
    const int lane = threadIdx.x & 31;
    const int half = lane >> 4;
    const int m    = lane & 15;
    const bf16_t* row = base + (size_t)m * ldr;
    v16bf f;
#pragma unroll
    for (int j = 0; j < 8; ++j) {
        const int kk = ((j < 4) ? 2 * j : 2 * j + 8) + 8 * half;
        f[2 * j]     = row[kk];
        f[2 * j + 1] = row[kk + 1];
    }
    return f;
}

// B-matrix fragment (32x16 bf16, KxN), stored K-contiguous per column:
// lane L holds column N=L%16; elements j<8 are K = j + 8*(L/16), j>=8 are K = j + 8 + 8*(L/16).
static __device__ inline v16bf load_b_frag(const bf16_t* __restrict__ base, int ldr) {
    const int lane = threadIdx.x & 31;
    const int half = lane >> 4;
    const int n    = lane & 15;
    const bf16_t* col = base + (size_t)n * ldr + 8 * half;
    v16bf f;
#pragma unroll
    for (int j = 0; j < 16; ++j) f[j] = col[(j < 8) ? j : j + 8];
    return f;
}

#define WMMA_BF16(a, b, c) \
    __builtin_amdgcn_wmma_f32_16x16x32_bf16(false, (a), false, (b), (short)0, (c), false, false)

// ---------------------------------------------------------------------------
// fp32 -> bf16 conversion
// ---------------------------------------------------------------------------
__global__ __launch_bounds__(256) void f32_to_bf16_kernel(const float* __restrict__ src,
                                                          bf16_t* __restrict__ dst, size_t n) {
    size_t i = (size_t)blockIdx.x * blockDim.x + threadIdx.x;
    const size_t stride = (size_t)gridDim.x * blockDim.x;
    for (; i < n; i += stride) dst[i] = (bf16_t)src[i];
}

// ---------------------------------------------------------------------------
// Tiled bf16 GEMM: C(MxN) = A(MxK) * B(KxN) (+bias), A/B row-major bf16.
// Block tile 128x128 (256 threads = 8 waves), each wave a 64x32 tile
// (4x2 WMMA 16x16 accumulators). A tile staged via async-to-LDS B128 DMA;
// B tile transposed via ds stores so B-frag K-runs are contiguous.
// Templated epilogue: no runtime branches in the store loop.
// ---------------------------------------------------------------------------
template <bool HAS_BIAS, bool OUT_F32>
__global__ __launch_bounds__(256) void gemm_bf16_kernel(
    const bf16_t* __restrict__ A, const bf16_t* __restrict__ Bm,
    const float* __restrict__ bias,
    bf16_t* __restrict__ Cb, float* __restrict__ Cf,
    int M, int N, int K)
{
    __shared__ bf16_t As[128][40];   // [m][k], 80B row stride (16B aligned)
    __shared__ bf16_t Bs[128][36];   // transposed: [n][k], padded

    const int tid  = threadIdx.x;
    const int lane = tid & 31;
    const int wave = tid >> 5;
    const int wm   = wave >> 2;      // 0..1  -> 64-row slab
    const int wn   = wave & 3;       // 0..3  -> 32-col slab
    const int mbase = blockIdx.y * 128;
    const int nbase = blockIdx.x * 128;

    v8f acc[4][2];
#pragma unroll
    for (int i = 0; i < 4; ++i)
#pragma unroll
        for (int j = 0; j < 2; ++j) acc[i][j] = zero_v8f();

    for (int kb = 0; kb < K; kb += 32) {
        {   // A tile 128x32: async DMA, 2 x B128 per thread (16B aligned)
            const int r  = tid >> 1;
            const int c0 = (tid & 1) * 16;
            const bf16_t* src = A + (size_t)(mbase + r) * K + kb + c0;
            async_load_b128(&As[r][c0],     src);
            async_load_b128(&As[r][c0 + 8], src + 8);
        }
        {   // B tile 32x128 -> Bs[n][k] transposed (element stores)
            const int kr = tid >> 3;             // 0..31
            const int c0 = (tid & 7) * 16;       // 0..112
            const bf16_t* src = Bm + (size_t)(kb + kr) * N + nbase + c0;
#pragma unroll
            for (int j = 0; j < 16; ++j) Bs[c0 + j][kr] = src[j];
        }
        wait_async0();
        __syncthreads();

        v16bf aF[4], bF[2];
#pragma unroll
        for (int ms = 0; ms < 4; ++ms) aF[ms] = load_a_frag(&As[wm * 64 + ms * 16][0], 40);
#pragma unroll
        for (int ns = 0; ns < 2; ++ns) bF[ns] = load_b_frag(&Bs[wn * 32 + ns * 16][0], 36);
#pragma unroll
        for (int ms = 0; ms < 4; ++ms)
#pragma unroll
            for (int ns = 0; ns < 2; ++ns)
                acc[ms][ns] = WMMA_BF16(aF[ms], bF[ns], acc[ms][ns]);
        __syncthreads();
    }

    // C layout: vgpr i, lane L -> row = i + 8*(L/16), col = L%16
    const int half = lane >> 4;
    const int ln   = lane & 15;
#pragma unroll
    for (int ms = 0; ms < 4; ++ms)
#pragma unroll
        for (int ns = 0; ns < 2; ++ns) {
            const int gm0 = mbase + wm * 64 + ms * 16 + 8 * half;
            const int gn  = nbase + wn * 32 + ns * 16 + ln;
            const float bv = HAS_BIAS ? bias[gn] : 0.0f;
            if (OUT_F32) {
                float* p = Cf + (size_t)gm0 * N + gn;
#pragma unroll
                for (int i = 0; i < 8; ++i) p[(size_t)i * N] = acc[ms][ns][i] + bv;
            } else {
                bf16_t* p = Cb + (size_t)gm0 * N + gn;
#pragma unroll
                for (int i = 0; i < 8; ++i) p[(size_t)i * N] = (bf16_t)(acc[ms][ns][i] + bv);
            }
        }
}

// ---------------------------------------------------------------------------
// Fused dual-stream flash attention.
// Grid: (T/64, B*H). Block: 128 threads (4 waves), wave w owns 16 query rows.
// K tile staged via async-to-LDS DMA; V tile transposed via ds stores;
// log_mask stream prefetched one s-tile ahead.
// ---------------------------------------------------------------------------
__global__ __launch_bounds__(128) void dual_attn_kernel(
    const bf16_t* __restrict__ qkv,      // [B*T][3*D]: q | k | v
    const float*  __restrict__ log_mask, // [H][T][T]
    bf16_t*       __restrict__ merged)   // [B*T][2*D]: out_txt | out_cau
{
    __shared__ bf16_t Ks[64][72];        // [s][d], 144B row stride (16B aligned)
    __shared__ bf16_t Vt[64][72];        // [d][s] (transposed for B-frags)
    __shared__ bf16_t Pt[4][16][72];     // per-wave P (txt), A-layout source
    __shared__ bf16_t Pc[4][16][72];     // per-wave P (cau)

    const int qtile = blockIdx.x;
    const int bh    = blockIdx.y;
    const int b     = bh / H_;
    const int h     = bh % H_;
    const int tid   = threadIdx.x;
    const int lane  = tid & 31;
    const int wave  = tid >> 5;
    const int half  = lane >> 4;
    const int ln    = lane & 15;

    const int qrow0 = qtile * 64 + wave * 16;

    // Q fragments for this wave's 16 rows (K = DH split into 2 x 32)
    v16bf qa[2];
#pragma unroll
    for (int ks = 0; ks < 2; ++ks)
        qa[ks] = load_a_frag(qkv + (size_t)(b * T_ + qrow0) * (3 * D_) + h * DH_ + ks * 32,
                             3 * D_);

    v8f acc_t[4], acc_c[4];
    float m_t[8], l_t[8], m_c[8], l_c[8];
#pragma unroll
    for (int ns = 0; ns < 4; ++ns) { acc_t[ns] = zero_v8f(); acc_c[ns] = zero_v8f(); }
#pragma unroll
    for (int i = 0; i < 8; ++i) { m_t[i] = -1e30f; l_t[i] = 0.f; m_c[i] = -1e30f; l_c[i] = 0.f; }

    const float scale = 0.125f;          // 1/sqrt(DH)

    for (int stile = 0; stile <= qtile; ++stile) {
        const int sbase = stile * 64;
        {   // K tile 64x64: async DMA, 4 x B128 per thread
            const int r  = tid >> 1;
            const int c0 = (tid & 1) * 32;
            const bf16_t* krow = qkv + (size_t)(b * T_ + sbase + r) * (3 * D_) + D_ + h * DH_ + c0;
#pragma unroll
            for (int u = 0; u < 4; ++u)
                async_load_b128(&Ks[r][c0 + 8 * u], krow + 8 * u);
            // V tile transposed (element-granular, ds stores)
            const bf16_t* vrow = qkv + (size_t)(b * T_ + sbase + r) * (3 * D_) + 2 * D_ + h * DH_ + c0;
#pragma unroll
            for (int j = 0; j < 32; ++j) Vt[c0 + j][r] = vrow[j];
        }
        // Prefetch next s-tile's log_mask block while this tile computes
        if (stile < qtile) {
            const int pr = tid >> 1;                 // 0..63 query rows of block
            const int pc = (tid & 1) * 32;           // 2 x 128B halves of row
            __builtin_prefetch(log_mask + ((size_t)h * T_ + qtile * 64 + pr) * T_
                                        + (sbase + 64) + pc, 0, 0);
        }
        wait_async0();
        __syncthreads();

        // ---- QK^T tile: 16 rows x 64 cols (4 subtiles), shared by streams
        float s_t[4][8], s_c[4][8];
#pragma unroll
        for (int st = 0; st < 4; ++st) {
            v8f c = zero_v8f();
#pragma unroll
            for (int ks = 0; ks < 2; ++ks) {
                v16bf bF = load_b_frag(&Ks[st * 16][ks * 32], 72);
                c = WMMA_BF16(qa[ks], bF, c);
            }
#pragma unroll
            for (int i = 0; i < 8; ++i) {
                const int qg = qrow0 + i + 8 * half;
                const int sg = sbase + st * 16 + ln;
                const float raw = c[i] * scale;
                const float lm  = log_mask[((size_t)h * T_ + qg) * T_ + sg];
                const float g   = 1.0f / (1.0f + __expf(-lm));
                float vt = raw, vc = raw * g;
                if (sg > qg) { vt = -1e30f; vc = -1e30f; }  // causal (diag tile only)
                s_t[st][i] = vt;
                s_c[st][i] = vc;
            }
        }

        // ---- online softmax update for both streams (row = i + 8*half,
        //      spread across 16 lanes of one half -> shfl width 16)
#pragma unroll
        for (int i = 0; i < 8; ++i) {
            {   // txt stream
                float rmax = s_t[0][i];
#pragma unroll
                for (int st = 1; st < 4; ++st) rmax = fmaxf(rmax, s_t[st][i]);
                for (int off = 1; off < 16; off <<= 1)
                    rmax = fmaxf(rmax, __shfl_xor(rmax, off, 16));
                const float mn   = fmaxf(m_t[i], rmax);
                const float corr = __expf(m_t[i] - mn);
                float rsum = 0.f;
#pragma unroll
                for (int st = 0; st < 4; ++st) {
                    const float p = __expf(s_t[st][i] - mn);
                    rsum += p;
                    Pt[wave][i + 8 * half][st * 16 + ln] = (bf16_t)p;
                }
                for (int off = 1; off < 16; off <<= 1)
                    rsum += __shfl_xor(rsum, off, 16);
                l_t[i] = l_t[i] * corr + rsum;
                m_t[i] = mn;
#pragma unroll
                for (int ns = 0; ns < 4; ++ns) acc_t[ns][i] = acc_t[ns][i] * corr;
            }
            {   // cau stream
                float rmax = s_c[0][i];
#pragma unroll
                for (int st = 1; st < 4; ++st) rmax = fmaxf(rmax, s_c[st][i]);
                for (int off = 1; off < 16; off <<= 1)
                    rmax = fmaxf(rmax, __shfl_xor(rmax, off, 16));
                const float mn   = fmaxf(m_c[i], rmax);
                const float corr = __expf(m_c[i] - mn);
                float rsum = 0.f;
#pragma unroll
                for (int st = 0; st < 4; ++st) {
                    const float p = __expf(s_c[st][i] - mn);
                    rsum += p;
                    Pc[wave][i + 8 * half][st * 16 + ln] = (bf16_t)p;
                }
                for (int off = 1; off < 16; off <<= 1)
                    rsum += __shfl_xor(rsum, off, 16);
                l_c[i] = l_c[i] * corr + rsum;
                m_c[i] = mn;
#pragma unroll
                for (int ns = 0; ns < 4; ++ns) acc_c[ns][i] = acc_c[ns][i] * corr;
            }
        }
        __syncthreads();   // P tiles visible; also orders Vt reuse below

        // ---- P @ V for both streams (K = s, 2 x 32 steps; N = DH, 4 subtiles)
#pragma unroll
        for (int ns = 0; ns < 4; ++ns) {
#pragma unroll
            for (int ks = 0; ks < 2; ++ks) {
                v16bf aT = load_a_frag(&Pt[wave][0][ks * 32], 72);
                v16bf aC = load_a_frag(&Pc[wave][0][ks * 32], 72);
                v16bf bF = load_b_frag(&Vt[ns * 16][ks * 32], 72);
                acc_t[ns] = WMMA_BF16(aT, bF, acc_t[ns]);
                acc_c[ns] = WMMA_BF16(aC, bF, acc_c[ns]);
            }
        }
        __syncthreads();   // done with Ks/Vt for this s-tile
    }

    // ---- epilogue: normalize and write merged = [out_txt | out_cau] (bf16)
#pragma unroll
    for (int ns = 0; ns < 4; ++ns)
#pragma unroll
        for (int i = 0; i < 8; ++i) {
            const int qg  = qrow0 + i + 8 * half;
            const int col = h * DH_ + ns * 16 + ln;
            const size_t rowoff = (size_t)(b * T_ + qg) * (2 * D_);
            merged[rowoff + col]      = (bf16_t)(acc_t[ns][i] / l_t[i]);
            merged[rowoff + D_ + col] = (bf16_t)(acc_c[ns][i] / l_c[i]);
        }
}

// ---------------------------------------------------------------------------
// Regularizer: lambda * mean(sigmoid(log_mask)) — deterministic 2-stage sum.
// ---------------------------------------------------------------------------
__global__ __launch_bounds__(256) void sigmoid_sum_partial_kernel(
    const float* __restrict__ lm, float* __restrict__ partial, size_t n)
{
    __shared__ float red[256];
    size_t i = (size_t)blockIdx.x * blockDim.x + threadIdx.x;
    const size_t stride = (size_t)gridDim.x * blockDim.x;
    float s = 0.f;
    for (; i < n; i += stride) s += 1.0f / (1.0f + __expf(-lm[i]));
    red[threadIdx.x] = s;
    __syncthreads();
    for (int o = 128; o > 0; o >>= 1) {
        if ((int)threadIdx.x < o) red[threadIdx.x] += red[threadIdx.x + o];
        __syncthreads();
    }
    if (threadIdx.x == 0) partial[blockIdx.x] = red[0];
}

__global__ __launch_bounds__(256) void finalize_reg_kernel(
    const float* __restrict__ partial, float* __restrict__ out, int nparts, float scale)
{
    __shared__ float red[256];
    float s = 0.f;
    for (int i = threadIdx.x; i < nparts; i += 256) s += partial[i];
    red[threadIdx.x] = s;
    __syncthreads();
    for (int o = 128; o > 0; o >>= 1) {
        if ((int)threadIdx.x < o) red[threadIdx.x] += red[threadIdx.x + o];
        __syncthreads();
    }
    if (threadIdx.x == 0) out[0] = red[0] * scale;
}

// ---------------------------------------------------------------------------
// Host launch
// ---------------------------------------------------------------------------
extern "C" void kernel_launch(void* const* d_in, const int* in_sizes, int n_in,
                              void* d_out, int out_size, void* d_ws, size_t ws_size,
                              hipStream_t stream)
{
    (void)in_sizes; (void)n_in; (void)out_size; (void)ws_size;

    const float* x        = (const float*)d_in[0];
    /* d_in[1] = causal_mask (bool) — causality computed analytically */
    const float* Wqkv     = (const float*)d_in[2];
    const float* log_mask = (const float*)d_in[3];
    const float* Wgate    = (const float*)d_in[4];
    const float* bgate    = (const float*)d_in[5];
    const float* Wout     = (const float*)d_in[6];
    float*       out      = (float*)d_out;

    char* ws = (char*)d_ws;
    size_t off = 0;
    auto alloc = [&](size_t bytes) -> void* {
        void* p = ws + off;
        off += (bytes + 255) & ~(size_t)255;
        return p;
    };

    const size_t nX    = (size_t)B_ * T_ * D_;       // 4,194,304
    bf16_t* xb      = (bf16_t*)alloc(nX * 2);
    bf16_t* wqkvb   = (bf16_t*)alloc((size_t)D_ * 3 * D_ * 2);
    bf16_t* wgb     = (bf16_t*)alloc((size_t)2 * D_ * D_ * 2);
    bf16_t* wob     = (bf16_t*)alloc((size_t)D_ * D_ * 2);
    bf16_t* qkvb    = (bf16_t*)alloc((size_t)B_ * T_ * 3 * D_ * 2);
    bf16_t* mergedb = (bf16_t*)alloc((size_t)B_ * T_ * 2 * D_ * 2);
    bf16_t* hiddenb = (bf16_t*)alloc((size_t)B_ * T_ * D_ * 2);
    float*  partials = (float*)alloc(1024 * sizeof(float));

    // 1) fp32 -> bf16 conversions
    f32_to_bf16_kernel<<<512, 256, 0, stream>>>(x, xb, nX);
    f32_to_bf16_kernel<<<512, 256, 0, stream>>>(Wqkv, wqkvb, (size_t)D_ * 3 * D_);
    f32_to_bf16_kernel<<<512, 256, 0, stream>>>(Wgate, wgb, (size_t)2 * D_ * D_);
    f32_to_bf16_kernel<<<512, 256, 0, stream>>>(Wout, wob, (size_t)D_ * D_);

    const int M = B_ * T_;   // 4096

    // 2) QKV projection: (4096 x 1024) @ (1024 x 3072) -> bf16
    gemm_bf16_kernel<false, false><<<dim3(3 * D_ / 128, M / 128), 256, 0, stream>>>(
        xb, wqkvb, nullptr, qkvb, nullptr, M, 3 * D_, D_);

    // 3) Fused dual-stream flash attention -> merged (4096 x 2048) bf16
    dual_attn_kernel<<<dim3(T_ / 64, B_ * H_), 128, 0, stream>>>(qkvb, log_mask, mergedb);

    // 4) Gate: (4096 x 2048) @ (2048 x 1024) + bgate -> bf16
    gemm_bf16_kernel<true, false><<<dim3(D_ / 128, M / 128), 256, 0, stream>>>(
        mergedb, wgb, bgate, hiddenb, nullptr, M, D_, 2 * D_);

    // 5) Output projection: (4096 x 1024) @ (1024 x 1024) -> fp32 d_out
    gemm_bf16_kernel<false, true><<<dim3(D_ / 128, M / 128), 256, 0, stream>>>(
        hiddenb, wob, nullptr, nullptr, out, M, D_, D_);

    // 6) reg scalar into d_out[B*T*D]
    const size_t nLM = (size_t)H_ * T_ * T_;
    sigmoid_sum_partial_kernel<<<1024, 256, 0, stream>>>(log_mask, partials, nLM);
    finalize_reg_kernel<<<1, 256, 0, stream>>>(partials, out + nX, 1024,
                                               0.0005f / (float)nLM);
}
